// HyperedgeEmbeddingExtractor_48799418417344
// MI455X (gfx1250) — matrix operand order repaired and checked by reading, not compiled, not run on hardware
//
#include <hip/hip_runtime.h>
#include <hip/hip_bf16.h>

typedef __attribute__((ext_vector_type(16))) _Float16 v16h;
typedef __attribute__((ext_vector_type(8)))  _Float16 v8h;
typedef __attribute__((ext_vector_type(4)))  _Float16 v4h;
typedef __attribute__((ext_vector_type(8)))  float    v8f;

#define D_DIM 512
#define LDSTRIDE 520   // 512 + 8 halves padding -> conflict-free ds_load_b128
#define MROWS 32       // rows per workgroup (2 WMMA M-tiles)

// ---------------------------------------------------------------------------
// Kernel 1: CSR offsets from sorted seg_ids via lower_bound binary search.
// ---------------------------------------------------------------------------
__global__ __launch_bounds__(256) void seg_offsets_kernel(
    const int* __restrict__ seg, int M, int E, int* __restrict__ off) {
  int e = blockIdx.x * blockDim.x + threadIdx.x;
  if (e > E) return;
  int lo = 0, hi = M;
  while (lo < hi) {
    int mid = (lo + hi) >> 1;
    if (seg[mid] < e) lo = mid + 1; else hi = mid;
  }
  off[e] = lo;
}

// ---------------------------------------------------------------------------
// Kernel 2: segment mean. One wave32 per edge; lane accumulates 4 float4
// columns; member loop unrolled x2 for memory-level parallelism.
// Result stored as f16 (feeds WMMA A operand).
// ---------------------------------------------------------------------------
__global__ __launch_bounds__(256) void seg_mean_kernel(
    const float* __restrict__ z, const int* __restrict__ midx,
    const int* __restrict__ off, int E, _Float16* __restrict__ meanbuf) {
  int gtid = blockIdx.x * blockDim.x + threadIdx.x;
  int wave = gtid >> 5;
  int lane = threadIdx.x & 31;
  if (wave >= E) return;

  int s = off[wave];
  int e = off[wave + 1];

  float4 acc[4];
#pragma unroll
  for (int q = 0; q < 4; ++q) acc[q] = make_float4(0.f, 0.f, 0.f, 0.f);

  int m = s;
  for (; m + 2 <= e; m += 2) {
    int idx0 = midx[m];
    int idx1 = midx[m + 1];
    const float4* row0 = (const float4*)(z + (size_t)idx0 * D_DIM);
    const float4* row1 = (const float4*)(z + (size_t)idx1 * D_DIM);
#pragma unroll
    for (int q = 0; q < 4; ++q) {
      float4 a = row0[lane + 32 * q];
      float4 b = row1[lane + 32 * q];
      acc[q].x += a.x + b.x; acc[q].y += a.y + b.y;
      acc[q].z += a.z + b.z; acc[q].w += a.w + b.w;
    }
  }
  if (m < e) {
    int idx = midx[m];
    const float4* row = (const float4*)(z + (size_t)idx * D_DIM);
#pragma unroll
    for (int q = 0; q < 4; ++q) {
      float4 v = row[lane + 32 * q];
      acc[q].x += v.x; acc[q].y += v.y; acc[q].z += v.z; acc[q].w += v.w;
    }
  }

  float inv = 1.0f / fmaxf((float)(e - s), 1.0f);
  v4h* dst = (v4h*)(meanbuf + (size_t)wave * D_DIM);
#pragma unroll
  for (int q = 0; q < 4; ++q) {
    v4h h;
    h[0] = (_Float16)(acc[q].x * inv);
    h[1] = (_Float16)(acc[q].y * inv);
    h[2] = (_Float16)(acc[q].z * inv);
    h[3] = (_Float16)(acc[q].w * inv);
    dst[lane + 32 * q] = h;
  }
}

// ---------------------------------------------------------------------------
// Kernel 3: pack W (f32 row-major [K=512][N=512]) into WMMA B-fragment order:
// pw[((ct*16 + kb)*32 + lane)*16 + e] = W[kb*32 + (e&7) + 8*(lane>>4) + 16*(e>>3)][ct*16 + (lane&15)]
// so each lane's per-K-step B fragment is 16 contiguous halves (32B).
// ---------------------------------------------------------------------------
__global__ __launch_bounds__(256) void pack_w_kernel(
    const float* __restrict__ W, _Float16* __restrict__ pw) {
  int t = blockIdx.x * blockDim.x + threadIdx.x;
  if (t >= D_DIM * D_DIM) return;
  int e  = t & 15;
  int l  = (t >> 4) & 31;
  int kb = (t >> 9) & 15;
  int ct = t >> 13;
  int col = ct * 16 + (l & 15);
  int K   = kb * 32 + (e & 7) + 8 * (l >> 4) + 16 * (e >> 3);
  pw[t] = (_Float16)W[(size_t)K * D_DIM + col];
}

// ---------------------------------------------------------------------------
// Kernel 4: fused GEMM (WMMA f32 <- f16xf16) + bias + LayerNorm + exact GELU.
// Block = 256 threads (8 waves), 32-row strip per block.
// Wave w owns column tiles ct = 4*w..4*w+3 across both 16-row M-tiles.
// B fragments indexed with 32-bit element offsets from the uniform pw base
// (SGPR-base + VGPR-offset addressing; unrolled deltas fold into IOFFSET).
// ---------------------------------------------------------------------------
__global__ __launch_bounds__(256) void gemm_ln_gelu_kernel(
    const _Float16* __restrict__ meanbuf, const _Float16* __restrict__ pw,
    const float* __restrict__ bias, const float* __restrict__ gamma,
    const float* __restrict__ beta, float* __restrict__ out, int E) {
  __shared__ _Float16 As[MROWS * LDSTRIDE];  // f16 A strip, padded (~33KB)
  __shared__ float    Hs[MROWS * LDSTRIDE];  // f32 pre-LN activations (~66KB)

  const int tid  = threadIdx.x;
  const int wave = tid >> 5;
  const int lane = tid & 31;
  const int r0   = blockIdx.x * MROWS;

  // ---- Stage the 32x512 f16 A strip into LDS (thread t: row t>>3, 64 cols)
  {
    int row   = tid >> 3;
    int cbase = (tid & 7) * 64;
    int grow  = r0 + row;
#pragma unroll
    for (int q = 0; q < 8; ++q) {
      v8h v = {};
      if (grow < E) {
        v = *(const v8h*)(meanbuf + (size_t)grow * D_DIM + cbase + 8 * q);
      }
      *(v8h*)(As + row * LDSTRIDE + cbase + 8 * q) = v;
    }
  }
  __syncthreads();

  // ---- WMMA main loop: 16 K-steps of 32; 2 M-tiles x 4 N-tiles per wave
  v8f acc[2][4] = {};
  const int hsel = lane >> 4;   // half-wave select
  const int arow = lane & 15;   // A row within M-tile
  const v16h* __restrict__ pwv = (const v16h*)pw;  // 32B fragments

#pragma unroll 2
  for (int kb = 0; kb < 16; ++kb) {
    // A fragments from LDS: two aligned 16B runs per lane per M-tile
    union { v16h v; v8h h[2]; } af[2];
#pragma unroll
    for (int mt = 0; mt < 2; ++mt) {
      const _Float16* ap = As + (arow + mt * 16) * LDSTRIDE + kb * 32 + 8 * hsel;
      af[mt].h[0] = *(const v8h*)(ap);
      af[mt].h[1] = *(const v8h*)(ap + 16);
    }
#pragma unroll
    for (int t = 0; t < 4; ++t) {
      int ct = wave * 4 + t;
      unsigned bidx = (unsigned)((ct * 16 + kb) * 32 + lane);  // v16h units
      v16h bf = pwv[bidx];
#pragma unroll
      for (int mt = 0; mt < 2; ++mt) {
        acc[mt][t] = __builtin_amdgcn_wmma_f32_16x16x32_f16(
            /*neg_a=*/false, af[mt].v, /*neg_b=*/false, bf,
            /*c_mod=*/(short)0, acc[mt][t],
            /*reuse_a=*/false, /*reuse_b=*/false);
      }
    }
  }

  // ---- Bias add, scatter C tiles into the f32 LDS strip
#pragma unroll
  for (int t = 0; t < 4; ++t) {
    int col = (wave * 4 + t) * 16 + (lane & 15);
    float bv = bias[col];
#pragma unroll
    for (int mt = 0; mt < 2; ++mt) {
#pragma unroll
      for (int r = 0; r < 8; ++r) {
        int row = mt * 16 + r + 8 * hsel;
        Hs[row * LDSTRIDE + col] = acc[mt][t][r] + bv;
      }
    }
  }
  __syncthreads();

  // ---- LayerNorm + exact GELU: wave w handles rows 4w..4w+3
  float g[16], be[16];
#pragma unroll
  for (int j = 0; j < 16; ++j) {
    int col = lane + 32 * j;
    g[j]  = gamma[col];
    be[j] = beta[col];
  }

#pragma unroll
  for (int rr = 0; rr < 4; ++rr) {
    int row  = wave * 4 + rr;
    int grow = r0 + row;

    float vals[16];
    float s = 0.f, ss = 0.f;
#pragma unroll
    for (int j = 0; j < 16; ++j) {
      float v = Hs[row * LDSTRIDE + lane + 32 * j];
      vals[j] = v;
      s  += v;
      ss += v * v;
    }
#pragma unroll
    for (int o = 16; o > 0; o >>= 1) {
      s  += __shfl_xor(s,  o, 32);
      ss += __shfl_xor(ss, o, 32);
    }
    const float invD = 1.0f / (float)D_DIM;
    float mu   = s * invD;
    float var  = ss * invD - mu * mu;
    float rstd = rsqrtf(var + 1e-5f);

    if (grow < E) {
#pragma unroll
      for (int j = 0; j < 16; ++j) {
        int col = lane + 32 * j;
        float x = (vals[j] - mu) * rstd * g[j] + be[j];
        out[(size_t)grow * D_DIM + col] =
            0.5f * x * (1.0f + erff(x * 0.70710678118654752f));
      }
    }
  }
}

// ---------------------------------------------------------------------------
extern "C" void kernel_launch(void* const* d_in, const int* in_sizes, int n_in,
                              void* d_out, int out_size, void* d_ws, size_t ws_size,
                              hipStream_t stream) {
  const float* z     = (const float*)d_in[0];   // [N,512]
  const float* W     = (const float*)d_in[1];   // [512,512]
  const float* bias  = (const float*)d_in[2];   // [512]
  const float* gamma = (const float*)d_in[3];   // [512]
  const float* beta  = (const float*)d_in[4];   // [512]
  const int*   midx  = (const int*)d_in[5];     // [M]
  const int*   seg   = (const int*)d_in[6];     // [M] sorted
  // d_in[7] = n_edges scalar (device only); derive E from out_size instead.

  const int M = in_sizes[5];
  const int E = out_size / D_DIM;

  // Workspace layout
  char* ws = (char*)d_ws;
  _Float16* meanbuf = (_Float16*)ws;                               // E*512 f16
  size_t o = (size_t)E * D_DIM * sizeof(_Float16);
  o = (o + 255) & ~(size_t)255;
  int* offs = (int*)(ws + o);                                      // (E+1) int
  o += (size_t)(E + 1) * sizeof(int);
  o = (o + 255) & ~(size_t)255;
  _Float16* pw = (_Float16*)(ws + o);                              // 512*512 f16
  (void)ws_size; (void)n_in;

  // 1) segment offsets
  {
    int nthr = E + 1;
    seg_offsets_kernel<<<(nthr + 255) / 256, 256, 0, stream>>>(seg, M, E, offs);
  }
  // 2) segment mean (one wave per edge)
  {
    int blocks = (E + 7) / 8;  // 8 waves per 256-thread block
    seg_mean_kernel<<<blocks, 256, 0, stream>>>(z, midx, offs, E, meanbuf);
  }
  // 3) pack W into WMMA B-fragment layout
  pack_w_kernel<<<(D_DIM * D_DIM) / 256, 256, 0, stream>>>(W, pw);
  // 4) fused WMMA GEMM + bias + LayerNorm + GELU
  {
    int blocks = (E + MROWS - 1) / MROWS;
    gemm_ln_gelu_kernel<<<blocks, 256, 0, stream>>>(meanbuf, pw, bias, gamma,
                                                    beta, (float*)d_out, E);
  }
}
